// RelationalNetwork_84782654423878
// MI455X (gfx1250) — matrix operand order repaired and checked by reading, not compiled, not run on hardware
//
#include <hip/hip_runtime.h>
#include <hip/hip_bf16.h>

// MI455X (gfx1250) relational GNN forward pass.
// bf16 activations/weights -> v_wmma_f32_16x16x32_bf16, f32 accumulate.
// GEMM: block = 64 rows x 256 cols (8 waves: 2M x 4N, wave = 32x64, 8 WMMA/K-step),
// double-buffered LDS A tile (one barrier per K-step), A read once from HBM for
// the H x H GEMMs (grid.y == 1). bf16 copies of residual streams are fused into
// the producing kernels to kill the per-layer conversion passes.
// Determinism: no float atomicAdd; BN via fixed-order partial-sum reduction;
// segment_max via uint atomicMax on nonnegative float bit patterns.

#define NN   20000
#define EE   320000
#define HH   256
#define FEA  64
#define LL   4
#define FLL  1024

typedef __attribute__((ext_vector_type(16))) __bf16 v16bf;
typedef __attribute__((ext_vector_type(8)))  __bf16 v8bf;
typedef __attribute__((ext_vector_type(8)))  float  v8f;

__device__ __forceinline__ unsigned short f2bf(float f) {
  unsigned u = __float_as_uint(f);
  return (unsigned short)((u + 0x7FFFu + ((u >> 16) & 1u)) >> 16);
}

// ---------------- f32 -> bf16 (round to nearest even) ----------------
__global__ void k_cvt_bf16(const float* __restrict__ in,
                           unsigned short* __restrict__ out, long n) {
  long i = (long)blockIdx.x * blockDim.x + threadIdx.x;
  long stride = (long)gridDim.x * blockDim.x;
  for (; i < n; i += stride) out[i] = f2bf(in[i]);
}

__global__ void k_zero_f32(float* __restrict__ p, long n) {
  long i = (long)blockIdx.x * blockDim.x + threadIdx.x;
  long stride = (long)gridDim.x * blockDim.x;
  for (; i < n; i += stride) p[i] = 0.0f;
}

// ---- bf16 WMMA GEMM: C[M,Nout] = A[M,K] * W[Nout,K]^T + bias (+ bf16 copy) ----
// Block 256 threads = 8 waves. Wave (wm = w>>2, wn = w&3): rows [mbase+32*wm,+32),
// cols [nbase+64*wn,+64). A tile 64x32 double-buffered in LDS; B straight from
// the L2-resident weight (one contiguous aligned 32B load per lane per fragment).
__global__ __launch_bounds__(256) void k_gemm_bf16(
    const unsigned short* __restrict__ A, const unsigned short* __restrict__ W,
    const float* __restrict__ bias, float* __restrict__ C,
    unsigned short* __restrict__ Cbf, int M, int K, int Nout) {
  __shared__ __align__(32) unsigned short As[2][64 * 40];

  const int mbase = blockIdx.x * 64;
  const int nbase = blockIdx.y * 256;
  const int wave  = threadIdx.x >> 5;
  const int lane  = threadIdx.x & 31;
  const int lhalf = lane >> 4;     // 0: lanes 0-15, 1: lanes 16-31
  const int l16   = lane & 15;
  const int wm    = wave >> 2;     // 0..1
  const int wn    = wave & 3;      // 0..3

  // cooperative A loader: thread -> one 8-half chunk (row r, col cc) of the tile
  const int r  = threadIdx.x >> 2;
  const int cc = (threadIdx.x & 3) << 3;
  int gr = mbase + r;
  gr = gr < M ? gr : M - 1;        // clamp: garbage rows never stored (guarded)
  const unsigned short* Arow = A + (size_t)gr * K + cc;
  unsigned short* const lds_dst0 = &As[0][r * 40 + cc];
  unsigned short* const lds_dst1 = &As[1][r * 40 + cc];

  // wave fragment pointers within a tile buffer
  const int frag_off = (wm * 32 + l16) * 40 + lhalf * 16;

  v8f acc0[4] = {}, acc1[4] = {};
  v8bf ld = *(const v8bf*)(Arow);

  for (int kb = 0; kb < K; kb += 32) {
    const int buf = (kb >> 5) & 1;
    *(v8bf*)(buf ? lds_dst1 : lds_dst0) = ld;
    __syncthreads();  // drains DScnt -> alternate-buffer WAR is ordered
    if (kb + 32 < K) {
      ld = *(const v8bf*)(Arow + kb + 32);
      if (kb + 64 < K)
        __builtin_prefetch((const void*)(Arow + kb + 64), 0, 1);
    }

    const unsigned short* ap0 = &As[buf][frag_off];
    const unsigned short* ap1 = ap0 + 16 * 40;
    v16bf av0 = __builtin_shufflevector(*(const v8bf*)ap0, *(const v8bf*)(ap0 + 8),
        0, 1, 2, 3, 4, 5, 6, 7, 8, 9, 10, 11, 12, 13, 14, 15);
    v16bf av1 = __builtin_shufflevector(*(const v8bf*)ap1, *(const v8bf*)(ap1 + 8),
        0, 1, 2, 3, 4, 5, 6, 7, 8, 9, 10, 11, 12, 13, 14, 15);

    const size_t wrbase = (size_t)(nbase + wn * 64 + l16) * K + kb + lhalf * 16;
#pragma unroll
    for (int nb = 0; nb < 4; ++nb) {
      v16bf bv = *(const v16bf*)(W + wrbase + (size_t)nb * 16 * K);
      acc0[nb] = __builtin_amdgcn_wmma_f32_16x16x32_bf16(false, av0, false, bv,
                                                         (short)0, acc0[nb], false, false);
      acc1[nb] = __builtin_amdgcn_wmma_f32_16x16x32_bf16(false, av1, false, bv,
                                                         (short)0, acc1[nb], false, false);
    }
  }

  // C/D layout: VGPR idx rr -> M = rr + 8*lhalf, N = l16
  const int crow0 = mbase + wm * 32 + lhalf * 8;
  const int ccol0 = nbase + wn * 64 + l16;
#pragma unroll
  for (int nb = 0; nb < 4; ++nb) {
    const float bnb = bias[ccol0 + nb * 16];
#pragma unroll
    for (int rr = 0; rr < 8; ++rr) {
      int row0 = crow0 + rr;
      if (row0 < M) {
        size_t idx = (size_t)row0 * Nout + ccol0 + nb * 16;
        float val = acc0[nb][rr] + bnb;
        C[idx] = val;
        if (Cbf) Cbf[idx] = f2bf(val);
      }
      int row1 = row0 + 16;
      if (row1 < M) {
        size_t idx = (size_t)row1 * Nout + ccol0 + nb * 16;
        float val = acc1[nb][rr] + bnb;
        C[idx] = val;
        if (Cbf) Cbf[idx] = f2bf(val);
      }
    }
  }
}

// ------- edge fuse pass 1: t = e + vp[src] + vp[dst]; per-block BN partials ---
__global__ __launch_bounds__(256) void k_edge_fuse1(
    float* __restrict__ t, const float* __restrict__ vp,
    const int* __restrict__ src, const int* __restrict__ dst,
    float* __restrict__ part, int rows) {
  const int c  = threadIdx.x;
  const int r0 = blockIdx.x * 128;
  float s = 0.0f, ss = 0.0f;
  for (int i = 0; i < 128; ++i) {
    int r = r0 + i;
    if (r >= rows) break;
    int sv = src[r], dv = dst[r];
    float v = t[(size_t)r * HH + c] + vp[(size_t)sv * HH + c] +
              vp[(size_t)dv * HH + c];
    t[(size_t)r * HH + c] = v;
    s += v;
    ss += v * v;
  }
  part[(size_t)blockIdx.x * 512 + c]       = s;
  part[(size_t)blockIdx.x * 512 + 256 + c] = ss;
}

// ------- deterministic stats reduce: mean + rstd per channel ------------------
__global__ void k_stats_reduce(const float* __restrict__ part, int nblk,
                               float inv_n, float* __restrict__ mr) {
  const int c = threadIdx.x;
  float s = 0.0f, ss = 0.0f;
  for (int b = 0; b < nblk; ++b) {
    s  += part[(size_t)b * 512 + c];
    ss += part[(size_t)b * 512 + 256 + c];
  }
  float m = s * inv_n;
  float var = ss * inv_n - m * m;
  mr[c]       = m;
  mr[256 + c] = rsqrtf(var + 1e-5f);
}

// -- edge apply: ep = relu(bn(t)); e_in += ep (f32 + fused bf16); scatter-max --
__global__ __launch_bounds__(256) void k_edge_apply(
    const float* __restrict__ t, const float* __restrict__ mr,
    const float* __restrict__ g, const float* __restrict__ b,
    float* __restrict__ e_in, unsigned short* __restrict__ e_bf,
    unsigned* __restrict__ vie,
    const int* __restrict__ src, const int* __restrict__ dst) {
  const int r = blockIdx.x;            // one edge row per block
  const int c = threadIdx.x;
  const size_t gid = (size_t)r * HH + c;
  float y = (t[gid] - mr[c]) * mr[256 + c] * g[c] + b[c];
  y = fmaxf(y, 0.0f);
  float e_new = e_in[gid] + y;
  e_in[gid] = e_new;
  e_bf[gid] = f2bf(e_new);
  unsigned bits = __float_as_uint(y);  // y >= 0: uint order == float order
  atomicMax(&vie[(size_t)src[r] * HH + c], bits);
  atomicMax(&vie[(size_t)dst[r] * HH + c], bits);
}

// ------- GRU elementwise + vertex BN partials ---------------------------------
__global__ __launch_bounds__(256) void k_gru_fused(
    const float* __restrict__ gi, const float* __restrict__ gh,
    const float* __restrict__ vin, float* __restrict__ h,
    float* __restrict__ part, int rows) {
  const int c  = threadIdx.x;
  const int r0 = blockIdx.x * 128;
  float s = 0.0f, ss = 0.0f;
  for (int i = 0; i < 128; ++i) {
    int r = r0 + i;
    if (r >= rows) break;
    size_t b3 = (size_t)r * (3 * HH);
    float ir = gi[b3 + c], iz = gi[b3 + 256 + c], ig = gi[b3 + 512 + c];
    float hr = gh[b3 + c], hz = gh[b3 + 256 + c], hg = gh[b3 + 512 + c];
    float rr = 1.0f / (1.0f + __expf(-(ir + hr)));
    float zz = 1.0f / (1.0f + __expf(-(iz + hz)));
    float ng = tanhf(ig + rr * hg);
    float hv = (1.0f - zz) * ng + zz * vin[(size_t)r * HH + c];
    h[(size_t)r * HH + c] = hv;
    s += hv;
    ss += hv * hv;
  }
  part[(size_t)blockIdx.x * 512 + c]       = s;
  part[(size_t)blockIdx.x * 512 + 256 + c] = ss;
}

// ------- vertex apply: v_in += relu(bn(h)) (f32 + fused bf16) -----------------
__global__ __launch_bounds__(256) void k_vert_apply(
    const float* __restrict__ h, const float* __restrict__ mr,
    const float* __restrict__ g, const float* __restrict__ b,
    float* __restrict__ v_in, unsigned short* __restrict__ v_bf) {
  const int c = threadIdx.x;
  const size_t gid = (size_t)blockIdx.x * HH + c;
  float y = (h[gid] - mr[c]) * mr[256 + c] * g[c] + b[c];
  float v_new = v_in[gid] + fmaxf(y, 0.0f);
  v_in[gid] = v_new;
  v_bf[gid] = f2bf(v_new);
}

// ------- collapse final two linears (FO==1): wcomb[h] = sum_f out_w[f]*fin_w[f,h]
__global__ void k_wcomb(const float* __restrict__ fin_w,
                        const float* __restrict__ fin_b,
                        const float* __restrict__ out_w,
                        const float* __restrict__ out_b,
                        float* __restrict__ wc) {
  const int hcol = threadIdx.x;
  float s = 0.0f;
  for (int f = 0; f < FLL; ++f) s += out_w[f] * fin_w[(size_t)f * HH + hcol];
  wc[hcol] = s;
  if (hcol == 0) {
    float bb = out_b[0];
    for (int f = 0; f < FLL; ++f) bb += out_w[f] * fin_b[f];
    wc[HH] = bb;
  }
}

// ------- final GEMV: out[e] = dot(e_in[e,:], wcomb) + bcomb -------------------
__global__ __launch_bounds__(256) void k_final(const float* __restrict__ e_in,
                                               const float* __restrict__ wc,
                                               float* __restrict__ out, int rows) {
  const int wave = threadIdx.x >> 5;
  const int lane = threadIdx.x & 31;
  const int r = blockIdx.x * 8 + wave;
  if (r >= rows) return;
  const float* row = e_in + (size_t)r * HH;
  float s = 0.0f;
#pragma unroll
  for (int j = 0; j < 8; ++j) {
    int c = lane + 32 * j;
    s += row[c] * wc[c];
  }
#pragma unroll
  for (int m = 16; m > 0; m >>= 1) s += __shfl_xor(s, m, 32);
  if (lane == 0) out[r] = s + wc[HH];
}

// =============================================================================
extern "C" void kernel_launch(void* const* d_in, const int* in_sizes, int n_in,
                              void* d_out, int out_size, void* d_ws, size_t ws_size,
                              hipStream_t stream) {
  const int N = NN, E = EE, H = HH, FE = FEA, FV = FEA, L = LL;
  const float* x         = (const float*)d_in[0];
  const float* edge_attr = (const float*)d_in[1];
  const int*   edge_idx  = (const int*)d_in[2];
  const float* ew  = (const float*)d_in[3];  const float* eb  = (const float*)d_in[4];
  const float* vw  = (const float*)d_in[5];  const float* vb  = (const float*)d_in[6];
  const float* Le_w = (const float*)d_in[7]; const float* Le_b = (const float*)d_in[8];
  const float* Lv_w = (const float*)d_in[9]; const float* Lv_b = (const float*)d_in[10];
  const float* bne_g = (const float*)d_in[11]; const float* bne_b = (const float*)d_in[12];
  const float* bnv_g = (const float*)d_in[13]; const float* bnv_b = (const float*)d_in[14];
  const float* gru_wih = (const float*)d_in[15]; const float* gru_whh = (const float*)d_in[16];
  const float* gru_bih = (const float*)d_in[17]; const float* gru_bhh = (const float*)d_in[18];
  const float* fin_w = (const float*)d_in[19]; const float* fin_b = (const float*)d_in[20];
  const float* out_w = (const float*)d_in[21]; const float* out_b = (const float*)d_in[22];
  const int* src = edge_idx;
  const int* dst = edge_idx + E;

  // ---- workspace carve ----
  char* ws = (char*)d_ws;
  size_t off = 0;
  auto carve = [&](size_t bytes) -> char* {
    char* p = ws + off;
    off = (off + bytes + 255) & ~(size_t)255;
    return p;
  };
  float* e_in = (float*)carve((size_t)E * H * 4);
  float* tmpE = (float*)carve((size_t)E * H * 4);
  float* v_in = (float*)carve((size_t)N * H * 4);
  float* vp   = (float*)carve((size_t)N * H * 4);   // reused as GRU h buffer
  float* vie  = (float*)carve((size_t)N * H * 4);
  float* gi   = (float*)carve((size_t)N * 3 * H * 4);
  float* gh   = (float*)carve((size_t)N * 3 * H * 4);
  unsigned short* Abf   = (unsigned short*)carve((size_t)E * H * 2);  // bf16 of e_in
  unsigned short* Vbf   = (unsigned short*)carve((size_t)N * H * 2);  // bf16 of v_in
  unsigned short* VIEbf = (unsigned short*)carve((size_t)N * H * 2);
  unsigned short* EAbf  = (unsigned short*)carve((size_t)E * FE * 2); // edge_attr bf16
  unsigned short* XBf   = (unsigned short*)carve((size_t)N * FV * 2); // x bf16
  unsigned short* Wew = (unsigned short*)carve((size_t)H * FE * 2);
  unsigned short* Wvw = (unsigned short*)carve((size_t)H * FV * 2);
  unsigned short* WLe = (unsigned short*)carve((size_t)L * H * H * 2);
  unsigned short* WLv = (unsigned short*)carve((size_t)L * H * H * 2);
  unsigned short* Wih = (unsigned short*)carve((size_t)L * 3 * H * H * 2);
  unsigned short* Whh = (unsigned short*)carve((size_t)L * 3 * H * H * 2);
  float* part = (float*)carve((size_t)2500 * 512 * 4);
  float* mr   = (float*)carve(2 * H * 4);
  float* wc   = (float*)carve((H + 1) * 4);

  auto cvt = [&](const float* s, unsigned short* d, long n) {
    k_cvt_bf16<<<2048, 256, 0, stream>>>(s, d, n);
  };
  auto gemm = [&](const unsigned short* A, const unsigned short* W,
                  const float* bias, float* C, unsigned short* Cbf,
                  int M, int K, int Nout) {
    dim3 g((M + 63) / 64, Nout / 256);
    k_gemm_bf16<<<g, 256, 0, stream>>>(A, W, bias, C, Cbf, M, K, Nout);
  };

  // weight + input conversion to bf16
  cvt(edge_attr, EAbf, (long)E * FE);
  cvt(x, XBf, (long)N * FV);
  cvt(ew, Wew, (long)H * FE);
  cvt(vw, Wvw, (long)H * FV);
  cvt(Le_w, WLe, (long)L * H * H);
  cvt(Lv_w, WLv, (long)L * H * H);
  cvt(gru_wih, Wih, (long)L * 3 * H * H);
  cvt(gru_whh, Whh, (long)L * 3 * H * H);

  // input projections (dual-store: f32 stream + bf16 copy for next GEMM)
  gemm(EAbf, Wew, eb, e_in, Abf, E, FE, H);
  gemm(XBf,  Wvw, vb, v_in, Vbf, N, FV, H);

  const int nblkE = (E + 127) / 128;  // 2500
  const int nblkN = (N + 127) / 128;  // 157

  for (int k = 0; k < L; ++k) {
    gemm(Abf, WLe + (size_t)k * H * H, Le_b + k * H, tmpE, nullptr, E, H, H);
    gemm(Vbf, WLv + (size_t)k * H * H, Lv_b + k * H, vp,   nullptr, N, H, H);

    k_edge_fuse1<<<nblkE, 256, 0, stream>>>(tmpE, vp, src, dst, part, E);
    k_stats_reduce<<<1, 256, 0, stream>>>(part, nblkE, 1.0f / (float)E, mr);
    k_zero_f32<<<1024, 256, 0, stream>>>(vie, (long)N * H);
    k_edge_apply<<<E, 256, 0, stream>>>(tmpE, mr, bne_g + k * H, bne_b + k * H,
                                        e_in, Abf, (unsigned*)vie, src, dst);

    cvt(vie, VIEbf, (long)N * H);
    gemm(VIEbf, Wih + (size_t)k * 3 * H * H, gru_bih + k * 3 * H, gi, nullptr, N, H, 3 * H);
    gemm(Vbf,   Whh + (size_t)k * 3 * H * H, gru_bhh + k * 3 * H, gh, nullptr, N, H, 3 * H);

    k_gru_fused<<<nblkN, 256, 0, stream>>>(gi, gh, v_in, vp, part, N);
    k_stats_reduce<<<1, 256, 0, stream>>>(part, nblkN, 1.0f / (float)N, mr);
    k_vert_apply<<<N, 256, 0, stream>>>(vp, mr, bnv_g + k * H, bnv_b + k * H,
                                        v_in, Vbf);
  }

  k_wcomb<<<1, 256, 0, stream>>>(fin_w, fin_b, out_w, out_b, wc);
  k_final<<<E / 8, 256, 0, stream>>>(e_in, wc, (float*)d_out, E);

  (void)in_sizes; (void)n_in; (void)out_size; (void)ws_size;
}
